// DVQVAE_34222299415029
// MI455X (gfx1250) — matrix operand order repaired and checked by reading, not compiled, not run on hardware
//
#include <hip/hip_runtime.h>

typedef unsigned short u16;
typedef __attribute__((ext_vector_type(16))) __bf16 v16bf;
typedef __attribute__((ext_vector_type(8)))  float  v8f;

union FragU {
    v16bf v;
    uint4 q[2];
    u16   u[16];
};

__device__ __forceinline__ u16 f2bf(float f) {
    unsigned u = __float_as_uint(f);
    u += 0x7FFFu + ((u >> 16) & 1u);   // round-to-nearest-even
    return (u16)(u >> 16);
}
__device__ __forceinline__ unsigned pack2(float a, float b) {
    return (unsigned)f2bf(a) | ((unsigned)f2bf(b) << 16);
}

// ---------------------------------------------------------------------------
// Tiled GEMM: Out[M,Nt] = act(A[M,K] @ Bw[K,Nt] + bias), bf16 WMMA, f32 acc.
// Block tile 256x64, 8 waves; wave w -> 32 rows x 64 cols (2x4 WMMA subtiles).
// Ping-pong LDS staging so global loads overlap WMMA.
// ---------------------------------------------------------------------------
template<bool A_F32, bool RELU, bool OUT_BF16>
__global__ __launch_bounds__(256)
void gemm_bias_kernel(const void* __restrict__ Av, const u16* __restrict__ Bw,
                      const float* __restrict__ bias, void* __restrict__ Outv,
                      int M, int Nt, int K)
{
    __shared__ struct {
        alignas(16) u16 A[2][256 * 40];  // 256 x 32, row stride 40 (80B, 16B-aligned)
        alignas(16) u16 B[2][64 * 40];   // B transposed: Bt[n][k], 64 x 32
    } sm;

    const int tid    = threadIdx.x;
    const int lane   = tid & 31;
    const int wave   = tid >> 5;
    const int lane16 = lane & 15;
    const int hb     = lane >> 4;            // lane half (0/1)
    const int Mbase  = blockIdx.y * 256;
    const int Nbase  = blockIdx.x * 64;
    const int arow0  = wave * 32 + lane16;   // first of this wave's two row groups

    v8f acc[2][4] = {};

    auto stage = [&](int p, int k0) {
        // --- A tile (256 x 32): one row per thread ---
        if (A_F32) {
            const float4* ap4 = (const float4*)((const float*)Av +
                                 (size_t)(Mbase + tid) * K + k0);
            unsigned* dst = (unsigned*)&sm.A[p][tid * 40];
#pragma unroll
            for (int j = 0; j < 8; ++j) {
                float4 f = ap4[j];
                dst[2 * j]     = pack2(f.x, f.y);
                dst[2 * j + 1] = pack2(f.z, f.w);
            }
        } else {
            const uint4* ap4 = (const uint4*)((const u16*)Av +
                                (size_t)(Mbase + tid) * K + k0);
            uint4* dst = (uint4*)&sm.A[p][tid * 40];
#pragma unroll
            for (int j = 0; j < 4; ++j) dst[j] = ap4[j];
        }
        // --- B tile transposed: Bt[n][k] = Bw[k0+k][Nbase+n] ---
        {
            const int kk = tid & 31;
            const int ng = (tid >> 5) * 8;
            const uint4 d = *(const uint4*)(Bw + (size_t)(k0 + kk) * Nt + Nbase + ng);
            const u16* dp = (const u16*)&d;
#pragma unroll
            for (int i = 0; i < 8; ++i) sm.B[p][(ng + i) * 40 + kk] = dp[i];
        }
    };

    int p = 0;
    stage(0, 0);

#pragma unroll 1
    for (int k0 = 0; k0 < K; k0 += 32) {
        __syncthreads();
        if (k0 + 32 < K) stage(p ^ 1, k0 + 32);

        // A fragments, contiguous K order: lanes<16 K0-15, lanes>=16 K16-31
        FragU af[2];
#pragma unroll
        for (int m = 0; m < 2; ++m) {
            const int ab = (arow0 + m * 16) * 40;
            af[m].q[0] = *(const uint4*)&sm.A[p][ab + hb * 16];
            af[m].q[1] = *(const uint4*)&sm.A[p][ab + hb * 16 + 8];
        }
#pragma unroll
        for (int s = 0; s < 4; ++s) {
            FragU bf;
            const int nb = (s * 16 + lane16) * 40 + hb * 16;
            bf.q[0] = *(const uint4*)&sm.B[p][nb];
            bf.q[1] = *(const uint4*)&sm.B[p][nb + 8];
#pragma unroll
            for (int m = 0; m < 2; ++m)
                acc[m][s] = __builtin_amdgcn_wmma_f32_16x16x32_bf16(
                    false, af[m].v, false, bf.v, (short)0, acc[m][s], false, false);
        }
        p ^= 1;
    }

    // epilogue: bias (+ReLU); C layout: VGPR r -> M = r + hb*8, N = lane16
#pragma unroll
    for (int s = 0; s < 4; ++s) {
        const int col = Nbase + s * 16 + lane16;
        const float bv = bias[col];
#pragma unroll
        for (int m = 0; m < 2; ++m) {
#pragma unroll
            for (int r = 0; r < 8; ++r) {
                const int row = Mbase + wave * 32 + m * 16 + r + hb * 8;
                float v = acc[m][s][r] + bv;
                if (RELU) v = fmaxf(v, 0.0f);
                if (OUT_BF16) ((u16*)Outv)[(size_t)row * Nt + col] = f2bf(v);
                else          ((float*)Outv)[(size_t)row * Nt + col] = v;
            }
        }
    }
}

// ---------------------------------------------------------------------------
// Fused score GEMM + argmin: argmin_k |e-c_k|^2 == argmax_k (e.c_k - 0.5|c_k|^2)
// 128 rows of Enc in 64KB LDS; A-fragments register-resident; codebook tiles
// streamed from L2 with double-buffered B fragments (loads overlap WMMA).
// Argmax update is branchless (v_cmp + v_cndmask, no EXEC juggling).
// ---------------------------------------------------------------------------
__device__ __forceinline__ void load_btile(FragU bf[8], const u16* __restrict__ CBb,
                                           int nt, int lane16, int hb)
{
    const u16* cb = CBb + (size_t)(nt * 16 + lane16) * 256 + hb * 16;
#pragma unroll
    for (int kk = 0; kk < 8; ++kk) {
        bf[kk].q[0] = *(const uint4*)(cb + kk * 32);
        bf[kk].q[1] = *(const uint4*)(cb + kk * 32 + 8);
    }
}

__device__ __forceinline__ void mm_tile(v8f& acc, const FragU af[8], const FragU bf[8])
{
#pragma unroll
    for (int kk = 0; kk < 8; ++kk)
        acc = __builtin_amdgcn_wmma_f32_16x16x32_bf16(
            false, af[kk].v, false, bf[kk].v, (short)0, acc, false, false);
}

__global__ __launch_bounds__(256)
void scores_argmin_kernel(const u16* __restrict__ Enc, const u16* __restrict__ CBb,
                          const float* __restrict__ halfsq, int* __restrict__ Idx)
{
    __shared__ alignas(16) u16 Et[128 * 256];   // exactly 64 KB

    const int tid    = threadIdx.x;
    const int lane   = tid & 31;
    const int wave   = tid >> 5;
    const int lane16 = lane & 15;
    const int hb     = lane >> 4;
    const int Mbase  = blockIdx.x * 128;

    {   // stage cols 0-63 and 128-191 of the 128x256 encoded tile
        const int row = tid >> 1;
        const int cg  = (tid & 1) * 128;
        const uint4* src = (const uint4*)(Enc + (size_t)(Mbase + row) * 256 + cg);
        uint4* dst = (uint4*)&Et[row * 256 + cg];
#pragma unroll
        for (int i = 0; i < 8; ++i) dst[i] = src[i];
    }
    __syncthreads();

    // all 8 A fragments (K=256) stay in registers
    FragU af[8];
    const int arow = wave * 16 + lane16;
#pragma unroll
    for (int kk = 0; kk < 8; ++kk) {
        af[kk].q[0] = *(const uint4*)&Et[arow * 256 + kk * 32 + hb * 16];
        af[kk].q[1] = *(const uint4*)&Et[arow * 256 + kk * 32 + hb * 16 + 8];
    }

    float bv[8];
    int   bi[8];
#pragma unroll
    for (int r = 0; r < 8; ++r) { bv[r] = -3.4e38f; bi[r] = 0; }

    auto process = [&](int nt, const v8f& acc) {
        const int col = nt * 16 + lane16;
        const float adj = halfsq[col];
#pragma unroll
        for (int r = 0; r < 8; ++r) {
            const float v = acc[r] - adj;
            // branchless: better = (v > best) | ((v == best) & (col < idx))
            const bool better = (v > bv[r]) | ((v == bv[r]) & (col < bi[r]));
            bv[r] = better ? v : bv[r];
            bi[r] = better ? col : bi[r];
        }
    };

    FragU b0[8], b1[8];
    load_btile(b0, CBb, 0, lane16, hb);

#pragma unroll 1
    for (int nt = 0; nt < 256; nt += 2) {           // 4096 codes, 16/tile, 2 tiles/iter
        load_btile(b1, CBb, nt + 1, lane16, hb);    // prefetch while computing b0
        {
            v8f acc = {};
            mm_tile(acc, af, b0);
            process(nt, acc);
        }
        load_btile(b0, CBb, (nt + 2 < 256) ? nt + 2 : 255, lane16, hb);
        {
            v8f acc = {};
            mm_tile(acc, af, b1);
            process(nt + 1, acc);
        }
    }

    // reduce across the 16-lane column groups (tie-break: lowest index), branchless
#pragma unroll
    for (int off = 8; off >= 1; off >>= 1) {
#pragma unroll
        for (int r = 0; r < 8; ++r) {
            const float ov = __shfl_xor(bv[r], off, 16);
            const int   oi = __shfl_xor(bi[r], off, 16);
            const bool better = (ov > bv[r]) | ((ov == bv[r]) & (oi < bi[r]));
            bv[r] = better ? ov : bv[r];
            bi[r] = better ? oi : bi[r];
        }
    }
    if (lane16 == 0) {
#pragma unroll
        for (int r = 0; r < 8; ++r)
            Idx[Mbase + wave * 16 + r + hb * 8] = bi[r];
    }
}

// ---------------------------------------------------------------------------
// helpers
// ---------------------------------------------------------------------------
__global__ void cvt_kernel(const float* __restrict__ in, u16* __restrict__ out, int n4)
{
    const int i = blockIdx.x * blockDim.x + threadIdx.x;
    if (i < n4) {
        float4 f = ((const float4*)in)[i];
        unsigned* o = (unsigned*)(out + (size_t)i * 4);
        o[0] = pack2(f.x, f.y);
        o[1] = pack2(f.z, f.w);
    }
}

__global__ void cb_halfsq_kernel(const float* __restrict__ CB, float* __restrict__ out)
{
    const int row  = blockIdx.x;
    const int lane = threadIdx.x;   // 32 threads
    float s = 0.0f;
#pragma unroll
    for (int i = 0; i < 8; ++i) {
        const float v = CB[(size_t)row * 256 + i * 32 + lane];
        s += v * v;
    }
#pragma unroll
    for (int off = 16; off >= 1; off >>= 1) s += __shfl_xor(s, off, 32);
    if (lane == 0) out[row] = 0.5f * s;
}

__global__ __launch_bounds__(256)
void gather_kernel(const int* __restrict__ Idx, const float* __restrict__ CB,
                   const u16* __restrict__ CBb, float* __restrict__ Qout,
                   u16* __restrict__ Qb)
{
    const int t    = threadIdx.x;
    const int row  = blockIdx.x * 8 + (t >> 5);
    const int lane = t & 31;
    const int k    = Idx[row];
    const float4* src = (const float4*)(CB + (size_t)k * 256);
    float4*       dst = (float4*)(Qout + (size_t)row * 256);
    dst[lane * 2]     = src[lane * 2];
    dst[lane * 2 + 1] = src[lane * 2 + 1];
    ((uint4*)(Qb + (size_t)row * 256))[lane] =
        ((const uint4*)(CBb + (size_t)k * 256))[lane];
}

// ---------------------------------------------------------------------------
extern "C" void kernel_launch(void* const* d_in, const int* in_sizes, int n_in,
                              void* d_out, int out_size, void* d_ws, size_t ws_size,
                              hipStream_t stream)
{
    const float* x  = (const float*)d_in[0];
    const float* W1 = (const float*)d_in[1];
    const float* b1 = (const float*)d_in[2];
    const float* W2 = (const float*)d_in[3];
    const float* b2 = (const float*)d_in[4];
    const float* CB = (const float*)d_in[5];
    const float* W3 = (const float*)d_in[6];
    const float* b3 = (const float*)d_in[7];
    const float* W4 = (const float*)d_in[8];
    const float* b4 = (const float*)d_in[9];

    const int N = 65536, IN = 1024, H = 2048, E = 256, KC = 4096;

    char* ws = (char*)d_ws;
    size_t off = 0;
    u16*   Wb1  = (u16*)(ws + off); off += (size_t)IN * H * 2;   // 4 MB
    u16*   Wb2  = (u16*)(ws + off); off += (size_t)H * E * 2;    // 1 MB
    u16*   Wb3  = (u16*)(ws + off); off += (size_t)E * H * 2;    // 1 MB
    u16*   Wb4  = (u16*)(ws + off); off += (size_t)H * IN * 2;   // 4 MB
    u16*   CBb  = (u16*)(ws + off); off += (size_t)KC * E * 2;   // 2 MB
    float* hsq  = (float*)(ws + off); off += (size_t)KC * 4;
    int*   idx  = (int*)(ws + off); off += (size_t)N * 4;
    u16*   enc  = (u16*)(ws + off); off += (size_t)N * E * 2;    // 32 MB
    u16*   qb   = (u16*)(ws + off); off += (size_t)N * E * 2;    // 32 MB
    u16*   hbuf = (u16*)(ws + off); off += (size_t)N * H * 2;    // 256 MB (h, reused as h2)

    // 1) fp32 -> bf16 weight / codebook conversion
    cvt_kernel<<<(IN * H / 4 + 255) / 256, 256, 0, stream>>>(W1, Wb1, IN * H / 4);
    cvt_kernel<<<(H * E / 4 + 255) / 256, 256, 0, stream>>>(W2, Wb2, H * E / 4);
    cvt_kernel<<<(E * H / 4 + 255) / 256, 256, 0, stream>>>(W3, Wb3, E * H / 4);
    cvt_kernel<<<(H * IN / 4 + 255) / 256, 256, 0, stream>>>(W4, Wb4, H * IN / 4);
    cvt_kernel<<<(KC * E / 4 + 255) / 256, 256, 0, stream>>>(CB, CBb, KC * E / 4);
    cb_halfsq_kernel<<<KC, 32, 0, stream>>>(CB, hsq);

    // 2) encoder
    gemm_bias_kernel<true, true, true>
        <<<dim3(H / 64, N / 256), 256, 0, stream>>>(x, Wb1, b1, hbuf, N, H, IN);
    gemm_bias_kernel<false, false, true>
        <<<dim3(E / 64, N / 256), 256, 0, stream>>>(hbuf, Wb2, b2, enc, N, E, H);

    // 3) nearest codebook entry (fused score GEMM + argmax)
    scores_argmin_kernel<<<N / 128, 256, 0, stream>>>(enc, CBb, hsq, idx);

    // 4) gather quantized (fp32 output, exact codebook rows) + bf16 copy
    gather_kernel<<<N / 8, 256, 0, stream>>>(idx, CB, CBb, (float*)d_out, qb);

    // 5) decoder
    gemm_bias_kernel<false, true, true>
        <<<dim3(H / 64, N / 256), 256, 0, stream>>>(qb, Wb3, b3, hbuf, N, H, E);
    gemm_bias_kernel<false, false, false>
        <<<dim3(IN / 64, N / 256), 256, 0, stream>>>(
            hbuf, Wb4, b4, (float*)d_out + (size_t)N * E, N, IN, H);
}
